// Kernel_transformer_21242908246411
// MI455X (gfx1250) — matrix operand order repaired and checked by reading, not compiled, not run on hardware
//
#include <hip/hip_runtime.h>
#include <hip/hip_bf16.h>
#include <math.h>

// ---------------- problem constants (from reference) ----------------
#define BB   4
#define SSQ  2048
#define DDIM 1024
#define HH   16
#define LLAY 6
#define CCLS 4
#define DHH  64
#define FFD  4096
#define BSD  (BB * SSQ)      // 8192 rows of activations
#define EPSV 1e-5f

typedef __bf16 bf16_t;
typedef __attribute__((ext_vector_type(16))) __bf16 v16bf;
typedef __attribute__((ext_vector_type(8)))  float  v8f;
typedef __attribute__((ext_vector_type(4)))  unsigned int u32x4;
typedef __attribute__((ext_vector_type(8)))  int i32x8;
typedef __attribute__((ext_vector_type(4)))  int i32x4;

#if defined(__has_builtin)
#  if __has_builtin(__builtin_amdgcn_tensor_load_to_lds)
#    define HAVE_TDM 1
#  endif
#endif
#ifndef HAVE_TDM
#  define HAVE_TDM 0
#endif

// ---------------- WMMA GEMM: Out = epi(A[M,K](bf16) @ W[K,N](f32->bf16)) ----
#define TILE_M 128
#define TILE_N 128
#define TILE_K 32
#define LDSK   48            // padded LDS row stride in halves (96B, 16B-aligned)

enum { EPI_F32 = 0, EPI_RELU_F32 = 1, EPI_RELU_BF16 = 2, EPI_RESIDUAL = 3 };

#if HAVE_TDM
// TDM: async-load one A tile (TILE_M rows x TILE_K bf16) into LDS with
// padding that reproduces the 48-half LDS row stride.
// pad_interval=3 -> pad every 16 DWORDs (=32 halves); pad_amount=7 -> 8 DWORDs (=16 halves).
__device__ inline void tdm_load_a_tile(unsigned int ldsAddr, const bf16_t* gptr,
                                       int M, int K) {
  const unsigned long long ga = (unsigned long long)(uintptr_t)gptr;
  u32x4 g0;
  g0[0] = 1u;                                            // count=1, user mode
  g0[1] = ldsAddr;                                       // LDS byte address
  g0[2] = (unsigned int)ga;                              // global_addr[31:0]
  g0[3] = (unsigned int)((ga >> 32) & 0x1FFFFFFu) | (2u << 30);  // addr[56:32] | type=2
  i32x8 g1;
  g1[0] = (1 << 16) | (1 << 20) | (3 << 22) | (7 << 25); // data_size=2B, pad_en, intvl, amt
  g1[1] = (int)(((unsigned)K & 0xFFFFu) << 16);          // tensor_dim0 lo (abar=0)
  g1[2] = (int)((((unsigned)K >> 16) & 0xFFFFu) | (((unsigned)M & 0xFFFFu) << 16));
  g1[3] = (int)((((unsigned)M >> 16) & 0xFFFFu) | ((unsigned)TILE_K << 16)); // tile_dim0=32
  g1[4] = TILE_M;                                        // tile_dim1=128, tile_dim2=0
  g1[5] = K;                                             // tensor_dim0_stride lo
  g1[6] = 0;
  g1[7] = 0;
  const i32x4 z4 = {0, 0, 0, 0};
#if __has_include(<hip/amd_detail/amd_gfx1250_TDM.h>)
  const i32x8 z8 = {0, 0, 0, 0, 0, 0, 0, 0};
  __builtin_amdgcn_tensor_load_to_lds(g0, g1, z4, z4, z8, 0);
#else
  __builtin_amdgcn_tensor_load_to_lds(g0, g1, z4, z4, 0);
#endif
}
#endif

// A fragment (16x32 bf16): lane L holds row L%16; K-halves per ISA table:
// lanes 0-15: K=0..7,16..23 ; lanes 16-31: K=8..15,24..31
__device__ inline v16bf frag_ld_a(const bf16_t* rowp, int lane) {
  const int base = (lane & 16) ? 8 : 0;
  union { uint4 q[2]; v16bf v; } u;
  u.q[0] = *reinterpret_cast<const uint4*>(rowp + base);
  u.q[1] = *reinterpret_cast<const uint4*>(rowp + 16 + base);
  return u.v;
}

// B fragment (32x16 bf16, staged K-transposed): lane L holds col L%16;
// lanes 0-15: K=0..7,16..23 ; lanes 16-31: K=8..15,24..31
__device__ inline v16bf frag_ld_b(const bf16_t* rowp, int lane) {
  const int base = (lane & 16) ? 8 : 0;
  union { uint4 q[2]; v16bf v; } u;
  u.q[0] = *reinterpret_cast<const uint4*>(rowp + base);
  u.q[1] = *reinterpret_cast<const uint4*>(rowp + 16 + base);
  return u.v;
}

template <int EPI>
__global__ __launch_bounds__(256) void gemm_wmma(
    const bf16_t* __restrict__ A, const float* __restrict__ W,
    int M, int N, int K,
    float* __restrict__ outF, bf16_t* __restrict__ outH,
    float* __restrict__ resF, bf16_t* __restrict__ resH,
    const float* __restrict__ rwPtr, int rwIdx)
{
  __shared__ __align__(16) bf16_t sA[2][TILE_M * LDSK];  // double-buffered (TDM)
  __shared__ __align__(16) bf16_t sB[TILE_N * LDSK];     // K-transposed: sB[n][k]

  const int t    = threadIdx.x;
  const int lane = t & 31;
  const int wid  = t >> 5;
  const int wr   = wid >> 1;            // 0..3  (M direction, 32 rows each)
  const int wc   = wid & 1;             // 0..1  (N direction, 64 cols each)
  const int bm   = blockIdx.y * TILE_M;
  const int bn   = blockIdx.x * TILE_N;

  const v8f vzero = {0.f, 0.f, 0.f, 0.f, 0.f, 0.f, 0.f, 0.f};
  v8f acc[2][4];
  #pragma unroll
  for (int tm = 0; tm < 2; ++tm)
    #pragma unroll
    for (int tn = 0; tn < 4; ++tn) acc[tm][tn] = vzero;

#if HAVE_TDM
  if (wid == 0)
    tdm_load_a_tile((unsigned int)(uintptr_t)&sA[0][0],
                    A + (size_t)bm * K, M, K);
#endif

  for (int k0 = 0; k0 < K; k0 += TILE_K) {
    const int cur = (k0 >> 5) & 1;

#if HAVE_TDM
    // prefetch next A tile into the other buffer via the Tensor Data Mover
    if (wid == 0 && (k0 + TILE_K) < K)
      tdm_load_a_tile((unsigned int)(uintptr_t)&sA[cur ^ 1][0],
                      A + (size_t)bm * K + (k0 + TILE_K), M, K);
#else
    // cooperative A staging: 128 rows x 32 halves, 16B chunks, static trip count
    #pragma unroll
    for (int it = 0; it < 2; ++it) {
      const int i = t + it * 256;
      const int r = i >> 2, c = (i & 3) * 8;
      *reinterpret_cast<uint4*>(&sA[cur][r * LDSK + c]) =
          *reinterpret_cast<const uint4*>(A + (size_t)(bm + r) * K + k0 + c);
    }
#endif

    // stage W tile transposed: sB[n][k] = bf16(W[k0+k][bn+n]).
    // Issue all 4 global loads first so they pipeline under one wait.
    float4 w4[4];
    #pragma unroll
    for (int it = 0; it < 4; ++it) {
      const int i = t + it * 256;           // 0..1023
      const int kk = i >> 5, nn = (i & 31) * 4;
      w4[it] = *reinterpret_cast<const float4*>(W + (size_t)(k0 + kk) * N + bn + nn);
    }
    #pragma unroll
    for (int it = 0; it < 4; ++it) {
      const int i = t + it * 256;
      const int kk = i >> 5, nn = (i & 31) * 4;
      sB[(nn + 0) * LDSK + kk] = (bf16_t)w4[it].x;
      sB[(nn + 1) * LDSK + kk] = (bf16_t)w4[it].y;
      sB[(nn + 2) * LDSK + kk] = (bf16_t)w4[it].z;
      sB[(nn + 3) * LDSK + kk] = (bf16_t)w4[it].w;
    }
    if ((k0 + TILE_K) < K)                  // prefetch next W tile (L2)
      __builtin_prefetch(W + (size_t)(k0 + TILE_K + (t >> 5)) * N + bn + (t & 31) * 4, 0, 1);

#if HAVE_TDM
    if (wid == 0) {
      if ((k0 + TILE_K) < K) __builtin_amdgcn_s_wait_tensorcnt(1);  // current tile done
      else                   __builtin_amdgcn_s_wait_tensorcnt(0);
    }
#endif
    __syncthreads();

    const int r16 = lane & 15;
    const bf16_t* aBase = &sA[cur][0];
    v16bf af[2], bfr[4];
    af[0] = frag_ld_a(aBase + (wr * 32 +      r16) * LDSK, lane);
    af[1] = frag_ld_a(aBase + (wr * 32 + 16 + r16) * LDSK, lane);
    #pragma unroll
    for (int tn = 0; tn < 4; ++tn)
      bfr[tn] = frag_ld_b(&sB[(wc * 64 + tn * 16 + r16) * LDSK], lane);

    #pragma unroll
    for (int tm = 0; tm < 2; ++tm)
      #pragma unroll
      for (int tn = 0; tn < 4; ++tn)
        acc[tm][tn] = __builtin_amdgcn_wmma_f32_16x16x32_bf16(
            false, af[tm], false, bfr[tn], (short)0, acc[tm][tn], false, false);
    __syncthreads();
  }

  // epilogue: C layout -> element e, lanes 0-15: M=e, lanes 16-31: M=e+8; N=lane%16
  float rw = 0.0f;
  if (EPI == EPI_RESIDUAL) rw = rwPtr[rwIdx];
  const int rbase = bm + wr * 32 + ((lane & 16) ? 8 : 0);
  const int cbase = bn + wc * 64 + (lane & 15);
  #pragma unroll
  for (int tm = 0; tm < 2; ++tm)
    #pragma unroll
    for (int tn = 0; tn < 4; ++tn)
      #pragma unroll
      for (int e = 0; e < 8; ++e) {
        const int gr = rbase + tm * 16 + e;
        const int gc = cbase + tn * 16;
        const size_t idx = (size_t)gr * N + gc;
        const float v = acc[tm][tn][e];
        if (EPI == EPI_F32) {
          outF[idx] = v;
        } else if (EPI == EPI_RELU_F32) {
          outF[idx] = fmaxf(v, 0.0f);
        } else if (EPI == EPI_RELU_BF16) {
          outH[idx] = (bf16_t)fmaxf(v, 0.0f);
        } else {  // EPI_RESIDUAL: x += rw*v, refresh bf16 copy
          const float nv = resF[idx] + rw * v;
          resF[idx] = nv;
          resH[idx] = (bf16_t)nv;
        }
      }
}

// ---------------- embedding gather (f32 + bf16 copies) ----------------------
__global__ __launch_bounds__(256) void embed_kernel(
    const int* __restrict__ ids, const float* __restrict__ emb,
    float* __restrict__ xF, bf16_t* __restrict__ xH)
{
  const size_t i = (size_t)blockIdx.x * 256 + threadIdx.x;   // over BSD*D
  const int tok = (int)(i >> 10);                            // D = 1024
  const int d   = (int)(i & 1023);
  const float v = emb[(size_t)ids[tok] * DDIM + d];
  xF[i] = v;
  xH[i] = (bf16_t)v;
}

// ---------------- cosFormer: kv = sum_s kf(s) (x) v(s), ksum = sum_s kf(s) ---
__global__ __launch_bounds__(256) void attn_kv_kernel(
    const float* __restrict__ kbuf, const float* __restrict__ vbuf,
    const int* __restrict__ amask, const int* __restrict__ lengths,
    float* __restrict__ kvOut, float* __restrict__ ksOut)
{
  const int bh = blockIdx.x;           // 0..63
  const int b = bh >> 4, h = bh & 15;
  __shared__ float sk[DHH];
  __shared__ float sv[DHH];
  const int t = threadIdx.x;
  const int kidx = t & 127;            // 0..127 (feature index, cos|sin halves)
  const int dh   = t >> 7;             // 0/1 -> which 32-wide V slice
  const float invLen = 1.0f / (float)lengths[b];

  float accKv[32];
  #pragma unroll
  for (int j = 0; j < 32; ++j) accKv[j] = 0.0f;
  float accSum = 0.0f;

  for (int s = 0; s < SSQ; ++s) {
    if (t < 128) {
      const size_t base = ((size_t)(b * SSQ + s)) * DDIM + h * DHH;
      if (t < 64) sk[t] = kbuf[base + t];
      else        sv[t - 64] = vbuf[base + (t - 64)];
    }
    __syncthreads();
    const float ang = 1.57079632679f * (float)s * invLen;
    const float wgt = (kidx < 64) ? __cosf(ang) : __sinf(ang);
    const float kf = wgt * (float)amask[b * SSQ + s] * sk[kidx & 63];
    accSum += kf;
    #pragma unroll
    for (int j = 0; j < 32; ++j) accKv[j] += kf * sv[dh * 32 + j];
    __syncthreads();
  }

  float* kvRow = kvOut + ((size_t)bh * 128 + kidx) * DHH + dh * 32;
  #pragma unroll
  for (int j = 0; j < 32; ++j) kvRow[j] = accKv[j];
  if (dh == 0) ksOut[bh * 128 + kidx] = accSum;
}

// ---------------- cosFormer: attn(s,d) = z_s * qf(s) . kv[:,d] --------------
#define SCHUNK 64
__global__ __launch_bounds__(256) void attn_out_kernel(
    const float* __restrict__ qbuf, const float* __restrict__ kvIn,
    const float* __restrict__ ksIn, const int* __restrict__ lengths,
    bf16_t* __restrict__ attnH)
{
  const int bh = blockIdx.x;
  const int b = bh >> 4, h = bh & 15;
  __shared__ float skv[128 * DHH];     // 32KB
  __shared__ float sks[128];
  __shared__ float sqf[4][128];

  const int t = threadIdx.x;
  for (int i = t; i < 128 * DHH; i += 256) skv[i] = kvIn[(size_t)bh * 128 * DHH + i];
  if (t < 128) sks[t] = ksIn[bh * 128 + t];
  __syncthreads();

  const int sl = t >> 6;               // 0..3 (s within group of 4)
  const int d  = t & 63;
  const float invLen = 1.0f / (float)lengths[b];
  const int s0base = blockIdx.y * SCHUNK;

  for (int s0 = s0base; s0 < s0base + SCHUNK; s0 += 4) {
    #pragma unroll
    for (int it = 0; it < 2; ++it) {       // stage 4 qf rows of 128 (static)
      const int i = t + it * 256;
      const int rs = i >> 7, kk = i & 127;
      const int ss = s0 + rs;
      const float ang = 1.57079632679f * (float)ss * invLen;
      const float wgt = (kk < 64) ? __cosf(ang) : __sinf(ang);
      sqf[rs][kk] = wgt * qbuf[((size_t)(b * SSQ + ss)) * DDIM + h * DHH + (kk & 63)];
    }
    __syncthreads();
    float num = 0.0f, den = 0.0f;
    #pragma unroll 8
    for (int kk = 0; kk < 128; ++kk) {
      const float qv = sqf[sl][kk];
      num += qv * skv[kk * DHH + d];
      den += qv * sks[kk];
    }
    const float val = num / (den + EPSV);
    attnH[((size_t)(b * SSQ + (s0 + sl))) * DDIM + h * DHH + d] = (bf16_t)val;
    __syncthreads();
  }
}

// ---------------- head: logits + CE loss ------------------------------------
__global__ __launch_bounds__(256) void head_kernel(
    const float* __restrict__ xF, const float* __restrict__ embOut,
    const int* __restrict__ labels, float* __restrict__ out)
{
  __shared__ float red[256];
  __shared__ float logits[BB * CCLS];
  const int t = threadIdx.x;
  const int pair = t >> 4;             // 0..15 -> (b,c)
  const int sub  = t & 15;
  const int b = pair >> 2, c = pair & 3;
  float p = 0.0f;
  for (int d = sub; d < DDIM; d += 16)
    p += xF[(size_t)(b * SSQ) * DDIM + d] * embOut[d * CCLS + c];
  red[t] = p;
  __syncthreads();
  for (int off = 8; off > 0; off >>= 1) {
    if (sub < off) red[t] += red[t + off];
    __syncthreads();
  }
  if (sub == 0) logits[pair] = red[t];
  __syncthreads();
  if (t == 0) {
    float loss = 0.0f;
    for (int bb = 0; bb < BB; ++bb) {
      float mx = -1e30f;
      for (int cc = 0; cc < CCLS; ++cc) mx = fmaxf(mx, logits[bb * CCLS + cc]);
      float se = 0.0f;
      for (int cc = 0; cc < CCLS; ++cc) se += __expf(logits[bb * CCLS + cc] - mx);
      const float lse = mx + __logf(se);
      loss += -(logits[bb * CCLS + labels[bb]] - lse);
    }
    out[0] = loss / (float)BB;
    for (int i = 0; i < BB * CCLS; ++i) out[1 + i] = logits[i];
  }
}

// ---------------- launch ----------------------------------------------------
extern "C" void kernel_launch(void* const* d_in, const int* in_sizes, int n_in,
                              void* d_out, int out_size, void* d_ws, size_t ws_size,
                              hipStream_t stream)
{
  const int*   input_ids = (const int*)d_in[0];
  const int*   labels    = (const int*)d_in[1];
  const int*   amask     = (const int*)d_in[2];
  const int*   lengths   = (const int*)d_in[3];
  const float* emb_in    = (const float*)d_in[4];
  const float* Wq        = (const float*)d_in[5];
  const float* Wk        = (const float*)d_in[6];
  const float* Wv        = (const float*)d_in[7];
  const float* Wo        = (const float*)d_in[8];
  const float* W1        = (const float*)d_in[9];
  const float* W2        = (const float*)d_in[10];
  const float* resw      = (const float*)d_in[11];
  const float* emb_out   = (const float*)d_in[12];
  float* out = (float*)d_out;

  char* w = (char*)d_ws;
  auto alloc = [&](size_t bytes) -> char* {
    char* p = w;
    w += (bytes + 255) & ~(size_t)255;
    return p;
  };
  float*  xF  = (float*) alloc((size_t)BSD * DDIM * 4);
  bf16_t* xH  = (bf16_t*)alloc((size_t)BSD * DDIM * 2);
  float*  qF  = (float*) alloc((size_t)BSD * DDIM * 4);
  float*  kF  = (float*) alloc((size_t)BSD * DDIM * 4);
  float*  vF  = (float*) alloc((size_t)BSD * DDIM * 4);
  bf16_t* aH  = (bf16_t*)alloc((size_t)BSD * DDIM * 2);
  bf16_t* h1H = (bf16_t*)alloc((size_t)BSD * FFD * 2);
  float*  kvW = (float*) alloc((size_t)BB * HH * 128 * DHH * 4);
  float*  ksW = (float*) alloc((size_t)BB * HH * 128 * 4);

  embed_kernel<<<(BSD * DDIM) / 256, 256, 0, stream>>>(input_ids, emb_in, xF, xH);

  const dim3 gD(DDIM / TILE_N, BSD / TILE_M);   // (8, 64)
  const dim3 gF(FFD  / TILE_N, BSD / TILE_M);   // (32, 64)

  for (int l = 0; l < LLAY; ++l) {
    const float* wq = Wq + (size_t)l * DDIM * DDIM;
    const float* wk = Wk + (size_t)l * DDIM * DDIM;
    const float* wv = Wv + (size_t)l * DDIM * DDIM;
    const float* wo = Wo + (size_t)l * DDIM * DDIM;
    const float* w1 = W1 + (size_t)l * DDIM * FFD;
    const float* w2 = W2 + (size_t)l * FFD * DDIM;

    gemm_wmma<EPI_RELU_F32><<<gD, 256, 0, stream>>>(
        xH, wq, BSD, DDIM, DDIM, qF, nullptr, nullptr, nullptr, nullptr, 0);
    gemm_wmma<EPI_RELU_F32><<<gD, 256, 0, stream>>>(
        xH, wk, BSD, DDIM, DDIM, kF, nullptr, nullptr, nullptr, nullptr, 0);
    gemm_wmma<EPI_F32><<<gD, 256, 0, stream>>>(
        xH, wv, BSD, DDIM, DDIM, vF, nullptr, nullptr, nullptr, nullptr, 0);

    attn_kv_kernel<<<BB * HH, 256, 0, stream>>>(kF, vF, amask, lengths, kvW, ksW);
    attn_out_kernel<<<dim3(BB * HH, SSQ / SCHUNK), 256, 0, stream>>>(
        qF, kvW, ksW, lengths, aH);

    gemm_wmma<EPI_RESIDUAL><<<gD, 256, 0, stream>>>(
        aH, wo, BSD, DDIM, DDIM, nullptr, nullptr, xF, xH, resw, l);
    gemm_wmma<EPI_RELU_BF16><<<gF, 256, 0, stream>>>(
        xH, w1, BSD, FFD, DDIM, nullptr, h1H, nullptr, nullptr, nullptr, 0);
    gemm_wmma<EPI_RESIDUAL><<<gD, 256, 0, stream>>>(
        h1H, w2, BSD, DDIM, FFD, nullptr, nullptr, xF, xH, resw, l);
  }

  head_kernel<<<1, 256, 0, stream>>>(xF, emb_out, labels, out);
}
